// MixedDecoder_up_858993459262
// MI455X (gfx1250) — matrix operand order repaired and checked by reading, not compiled, not run on hardware
//
#include <hip/hip_runtime.h>
#include <stdint.h>

// ---------------------------------------------------------------- constants
#define BATCH   1024
#define LAT     64
#define FCON    256
#define HID     512
#define NEXP    8
#define GATEH   64
#define OUTSZ   512
#define IN0     320          // LAT + FCON
#define IN12    576          // LAT + HID
#define KPAD0   2592         // 8*320 + 8 -> pad to mult of 32
#define KPAD12  4640         // 8*576 + 8 -> pad to mult of 32

typedef _Float16 f16;
typedef f16   v16h __attribute__((ext_vector_type(16)));
typedef float v8f  __attribute__((ext_vector_type(8)));

// ---------------------------------------------------------------- helpers
__device__ __forceinline__ uint16_t f2h(float f) {
    union { f16 h; uint16_t u; } cv;
    cv.h = (f16)f;
    return cv.u;
}

__device__ __forceinline__ float elu(float x) {
    return x > 0.f ? x : expm1f(x);
}

// A fragment: 16-bit A 16x32 layout (contiguous K halves per lane group).
// ptr is row base already offset by (half*8) dwords; kd = chunk base in dwords.
__device__ __forceinline__ v16h load_frag_a(const uint32_t* __restrict__ p, int kd) {
    union { v16h v; uint4 q[2]; } f;
    f.q[0] = *reinterpret_cast<const uint4*>(p + kd);
    f.q[1] = *reinterpret_cast<const uint4*>(p + kd + 4);
    return f.v;
}

// B fragment: 16-bit B 32x16 layout (contiguous K halves per lane group).
// ptr is column-row (Wt row) base already offset by (half*8) dwords.
__device__ __forceinline__ v16h load_frag_b(const uint32_t* __restrict__ p, int kd) {
    union { v16h v; uint4 q[2]; } f;
    f.q[0] = *reinterpret_cast<const uint4*>(p + kd);
    f.q[1] = *reinterpret_cast<const uint4*>(p + kd + 4);
    return f.v;
}

// ---------------------------------------------------------------- gate MLP
// one block (64 threads) per batch row
__global__ __launch_bounds__(64) void gate_kernel(
    const float* __restrict__ z,  const float* __restrict__ c,
    const float* __restrict__ gw1, const float* __restrict__ gb1,
    const float* __restrict__ gw2, const float* __restrict__ gb2,
    const float* __restrict__ gw3, const float* __restrict__ gb3,
    float* __restrict__ coeff)
{
    __shared__ float x[IN0];
    __shared__ float h1[GATEH];
    __shared__ float h2[GATEH];
    __shared__ float lg[NEXP];
    const int b = blockIdx.x, t = threadIdx.x;

    for (int i = t; i < IN0; i += 64)
        x[i] = (i < LAT) ? z[b * LAT + i] : c[b * FCON + (i - LAT)];
    __syncthreads();

    float s = gb1[t];
    for (int i = 0; i < IN0; ++i) s += x[i] * gw1[i * GATEH + t];
    h1[t] = elu(s);
    __syncthreads();

    s = gb2[t];
    for (int i = 0; i < GATEH; ++i) s += h1[i] * gw2[i * GATEH + t];
    h2[t] = elu(s);
    __syncthreads();

    if (t < NEXP) {
        s = gb3[t];
        for (int i = 0; i < GATEH; ++i) s += h2[i] * gw3[i * NEXP + t];
        lg[t] = s;
    }
    __syncthreads();

    if (t == 0) {
        float mx = lg[0];
        for (int j = 1; j < NEXP; ++j) mx = fmaxf(mx, lg[j]);
        float e[NEXP], sum = 0.f;
        for (int j = 0; j < NEXP; ++j) { e[j] = expf(lg[j] - mx); sum += e[j]; }
        for (int j = 0; j < NEXP; ++j) coeff[b * NEXP + j] = e[j] / sum;
    }
}

// ---------------------------------------------------------------- weight staging
// dst[o][k] f16, k = e*in + i stacks experts; bias appended as rows E*in..E*in+7; zero pad
__global__ __launch_bounds__(256) void prep_w_kernel(
    const float* __restrict__ w, const float* __restrict__ bias,
    uint16_t* __restrict__ dst, int in, int out, int Kpad)
{
    const int idx = blockIdx.x * 256 + threadIdx.x;
    if (idx >= out * Kpad) return;
    const int o = idx / Kpad;
    const int k = idx - o * Kpad;
    const int ein = NEXP * in;
    float v = 0.f;
    if (k < ein)            v = w[(size_t)k * out + o];        // w[(e*in+i)*out + o]
    else if (k < ein + NEXP) v = bias[(k - ein) * out + o];
    dst[(size_t)o * Kpad + k] = f2h(v);
}

// ---------------------------------------------------------------- activation staging
// dst[b][k] f16, k = e*in + i : coeff[b,e] * concat(z, aux)[b,i]; coeff appended; zero pad
__global__ __launch_bounds__(256) void prep_a_kernel(
    const float* __restrict__ z, const float* __restrict__ aux,
    const float* __restrict__ coeff, uint16_t* __restrict__ dst,
    int in, int auxW, int Kpad)
{
    const int idx = blockIdx.x * 256 + threadIdx.x;
    if (idx >= BATCH * Kpad) return;
    const int b = idx / Kpad;
    const int k = idx - b * Kpad;
    const int ein = NEXP * in;
    float v = 0.f;
    if (k < ein) {
        const int e = k / in;
        const int i = k - e * in;
        const float s = (i < LAT) ? z[b * LAT + i] : aux[(size_t)b * auxW + (i - LAT)];
        v = coeff[b * NEXP + e] * s;
    } else if (k < ein + NEXP) {
        v = coeff[b * NEXP + (k - ein)];
    }
    dst[(size_t)b * Kpad + k] = f2h(v);
}

// ---------------------------------------------------------------- WMMA GEMM
// out[M,N] = A[M,Kpad] * Wt[N,Kpad]^T , fp32 accumulate, optional ELU.
// One wave -> 16-row M tile x 4 N tiles (64 cols). Block = 8 waves.
__global__ __launch_bounds__(256) void moe_gemm_kernel(
    const uint32_t* __restrict__ A, const uint32_t* __restrict__ Wt,
    float* __restrict__ out, int M, int N, int Kpad, int applyElu)
{
    const int lane = threadIdx.x & 31;
    const int wave = threadIdx.x >> 5;
    const int NG   = N >> 6;                       // groups of 64 cols
    const int job  = blockIdx.x * 8 + wave;
    const int m_tile  = job / NG;
    const int n_group = job - m_tile * NG;
    if (m_tile * 16 >= M) return;

    const int half = lane >> 4;
    const int mn   = lane & 15;
    const int Kd   = Kpad >> 1;                    // dwords (f16 pairs) per row

    const uint32_t* aRow = A  + (size_t)(m_tile * 16 + mn) * Kd + half * 8;
    const uint32_t* bR0  = Wt + (size_t)(n_group * 64 + mn) * Kd + half * 8;
    const uint32_t* bR1  = bR0 + (size_t)16 * Kd;
    const uint32_t* bR2  = bR0 + (size_t)32 * Kd;
    const uint32_t* bR3  = bR0 + (size_t)48 * Kd;

    v8f acc0 = {}, acc1 = {}, acc2 = {}, acc3 = {};

    for (int kd = 0; kd < Kd; kd += 16) {          // 32 K elements per step
        const v16h a = load_frag_a(aRow, kd);
        v16h w;
        w = load_frag_b(bR0, kd);
        acc0 = __builtin_amdgcn_wmma_f32_16x16x32_f16(false, a, false, w, (short)0, acc0, false, false);
        w = load_frag_b(bR1, kd);
        acc1 = __builtin_amdgcn_wmma_f32_16x16x32_f16(false, a, false, w, (short)0, acc1, false, false);
        w = load_frag_b(bR2, kd);
        acc2 = __builtin_amdgcn_wmma_f32_16x16x32_f16(false, a, false, w, (short)0, acc2, false, false);
        w = load_frag_b(bR3, kd);
        acc3 = __builtin_amdgcn_wmma_f32_16x16x32_f16(false, a, false, w, (short)0, acc3, false, false);
    }

    // C/D layout: VGPR r, lanes 0-15 -> row r, col lane; lanes 16-31 -> row r+8
    const int rowBase = m_tile * 16 + half * 8;
    const int colBase = n_group * 64 + mn;
    for (int r = 0; r < 8; ++r) {
        const size_t ro = (size_t)(rowBase + r) * N + colBase;
        float v0 = acc0[r], v1 = acc1[r], v2 = acc2[r], v3 = acc3[r];
        if (applyElu) { v0 = elu(v0); v1 = elu(v1); v2 = elu(v2); v3 = elu(v3); }
        out[ro]      = v0;
        out[ro + 16] = v1;
        out[ro + 32] = v2;
        out[ro + 48] = v3;
    }
}

// ---------------------------------------------------------------- launch
extern "C" void kernel_launch(void* const* d_in, const int* in_sizes, int n_in,
                              void* d_out, int out_size, void* d_ws, size_t ws_size,
                              hipStream_t stream) {
    (void)in_sizes; (void)n_in; (void)out_size; (void)ws_size;

    const float* z   = (const float*)d_in[0];
    const float* c   = (const float*)d_in[1];
    const float* w0  = (const float*)d_in[2];
    const float* b0  = (const float*)d_in[3];
    const float* w1  = (const float*)d_in[4];
    const float* b1  = (const float*)d_in[5];
    const float* w2  = (const float*)d_in[6];
    const float* b2  = (const float*)d_in[7];
    const float* gw1 = (const float*)d_in[8];
    const float* gb1 = (const float*)d_in[9];
    const float* gw2 = (const float*)d_in[10];
    const float* gb2 = (const float*)d_in[11];
    const float* gw3 = (const float*)d_in[12];
    const float* gb3 = (const float*)d_in[13];
    float* out = (float*)d_out;

    // workspace layout (all offsets 256B aligned)
    char* ws = (char*)d_ws;
    float*    coeff = (float*)ws;                                  //  32 KB
    float*    h     = (float*)(ws + 32768);                        //   2 MB (layer out ping)
    uint16_t* Abuf  = (uint16_t*)(ws + 2129920);                   // 9.5 MB (staged A, reused)
    uint16_t* Wbuf  = (uint16_t*)(ws + 11632640);                  // 4.75 MB (staged Wt, reused)

    const int gemmBlocks = (BATCH / 16) * (OUTSZ / 64) / 8;        // 64

    // gate -> coeff
    gate_kernel<<<BATCH, 64, 0, stream>>>(z, c, gw1, gb1, gw2, gb2, gw3, gb3, coeff);

    // ---- layer 0: [1024 x 2592] x [2592 x 512], ELU
    prep_w_kernel<<<(OUTSZ * KPAD0 + 255) / 256, 256, 0, stream>>>(w0, b0, Wbuf, IN0, OUTSZ, KPAD0);
    prep_a_kernel<<<(BATCH * KPAD0 + 255) / 256, 256, 0, stream>>>(z, c, coeff, Abuf, IN0, FCON, KPAD0);
    moe_gemm_kernel<<<gemmBlocks, 256, 0, stream>>>((const uint32_t*)Abuf, (const uint32_t*)Wbuf,
                                                    h, BATCH, OUTSZ, KPAD0, 1);

    // ---- layer 1: [1024 x 4640] x [4640 x 512], ELU
    prep_w_kernel<<<(OUTSZ * KPAD12 + 255) / 256, 256, 0, stream>>>(w1, b1, Wbuf, IN12, OUTSZ, KPAD12);
    prep_a_kernel<<<(BATCH * KPAD12 + 255) / 256, 256, 0, stream>>>(z, h, coeff, Abuf, IN12, HID, KPAD12);
    moe_gemm_kernel<<<gemmBlocks, 256, 0, stream>>>((const uint32_t*)Abuf, (const uint32_t*)Wbuf,
                                                    h, BATCH, OUTSZ, KPAD12, 1);

    // ---- layer 2: [1024 x 4640] x [4640 x 512], no act -> d_out
    prep_w_kernel<<<(OUTSZ * KPAD12 + 255) / 256, 256, 0, stream>>>(w2, b2, Wbuf, IN12, OUTSZ, KPAD12);
    prep_a_kernel<<<(BATCH * KPAD12 + 255) / 256, 256, 0, stream>>>(z, h, coeff, Abuf, IN12, HID, KPAD12);
    moe_gemm_kernel<<<gemmBlocks, 256, 0, stream>>>((const uint32_t*)Abuf, (const uint32_t*)Wbuf,
                                                    out, BATCH, OUTSZ, KPAD12, 0);
}
